// Model_60662118088790
// MI455X (gfx1250) — compile-verified, matrix operand order checked by reading, not run on hardware
//
#include <hip/hip_runtime.h>
#include <hip/hip_bf16.h>

// ---------------------------------------------------------------------------
// CDNA5 (gfx1250) implementation.
//  * All 1x1-conv GEMMs use v_wmma_f32_16x16x32_f16 (wave32, fp32 accum).
//    Block tile 32(M) x 128(N), 8 waves; each wave computes two 16x16 tiles
//    (B fragment reused against two A tiles) -> 2 WMMAs per K-step.
//  * LDS double-buffered: stage tile k+1 while computing tile k; one barrier
//    per K-step. X tile stored transposed so A and B fragments are each two
//    contiguous 16B runs per lane (ds_load_b128).
//  * Global staging: unconditional float4 loads (global_load_b128); stem K=6
//    zero-padded to 32 so the hot path is guard-free. global_prefetch_b8 for
//    the K+2 tile.
//  * Splat/gather/grid-conv are VALU + global float atomics.
// Internal activation layout: (C, BN) with BN = B*N = 16384, pt = b*N + n.
// ---------------------------------------------------------------------------

typedef __attribute__((ext_vector_type(16))) _Float16 v16h;
typedef __attribute__((ext_vector_type(8)))  _Float16 v8h;
typedef __attribute__((ext_vector_type(4)))  _Float16 v4h;
typedef __attribute__((ext_vector_type(8)))  float    v8f;

#define TILE_N 128
#define TILE_M 32
#define TILE_K 32
#define LDS_STRIDE 40   // halfs; 80 bytes: 16B aligned, conflict-free spacing

// ----------------------------- WMMA GEMM -----------------------------------
// Y[row, n] = epilogue( sum_k W[row,k] * X[k,n] )
//   K  : X K-extent (multiple of TILE_K; NP multiple of TILE_N)
//   Kw : actual W columns (= row stride of W). Kw <= K; rows k >= Kw are 0.
// epilogue: +bias, +wp@xyz (keys), BN, relu/tanh, +residual, optional (B,C,N)
// output layout for the final classifier.
__global__ __launch_bounds__(256)
void wmma_gemm(const float* __restrict__ W, const float* __restrict__ X,
               const float* __restrict__ bias,
               const float* __restrict__ bng, const float* __restrict__ bnb,
               const float* __restrict__ bnm, const float* __restrict__ bnv,
               const float* __restrict__ wp,  const float* __restrict__ xyzp,
               const float* __restrict__ resid,
               float* __restrict__ Y,
               int Cout, int K, int Kw, int NP, int act, int out_mode, int Nper)
{
    __shared__ _Float16 sxT[2][TILE_N][LDS_STRIDE];  // X tiles, transposed, f16
    __shared__ _Float16 sa [2][TILE_M][LDS_STRIDE];  // W tiles, f16

    const int tid   = threadIdx.x;
    const int lane  = tid & 31;
    const int wave  = tid >> 5;                   // 8 waves / block
    const int p0    = blockIdx.x * TILE_N;
    const int m0    = blockIdx.y * TILE_M;
    const int col   = lane & 15;
    const int khalf = (lane >> 4) * 8;            // ISA 16-bit A/B lane split

    // stage one (X,W) K-tile into LDS buffer dst
    auto stage = [&](int dst, int kb) {
        // X tile (TILE_K x TILE_N) transposed, float4 loads, guard-free
#pragma unroll
        for (int i = 0; i < 4; ++i) {
            int idx = tid + i * 256;              // 0..1023
            int kk  = idx >> 5;                   // 0..31
            int c4  = (idx & 31) << 2;            // 0..124
            const float4 v = *(const float4*)(X + (size_t)(kb + kk) * NP + p0 + c4);
            sxT[dst][c4 + 0][kk] = (_Float16)v.x;
            sxT[dst][c4 + 1][kk] = (_Float16)v.y;
            sxT[dst][c4 + 2][kk] = (_Float16)v.z;
            sxT[dst][c4 + 3][kk] = (_Float16)v.w;
        }
        // W tile (TILE_M x TILE_K)
        if (m0 + TILE_M <= Cout && kb + TILE_K <= Kw) {   // fast path (uniform)
            int mm = tid >> 3;                    // 0..31
            int k4 = (tid & 7) << 2;              // 0..28
            const float4 v = *(const float4*)(W + (size_t)(m0 + mm) * Kw + kb + k4);
            v4h h = { (_Float16)v.x, (_Float16)v.y, (_Float16)v.z, (_Float16)v.w };
            *(v4h*)&sa[dst][mm][k4] = h;
        } else {                                  // stem Kw=6 / Cout=48,13 tails
            for (int i = tid; i < TILE_M * TILE_K; i += 256) {
                int mm = i >> 5, kk = i & 31;
                float v = 0.f;
                if (m0 + mm < Cout && kb + kk < Kw)
                    v = W[(size_t)(m0 + mm) * Kw + kb + kk];
                sa[dst][mm][kk] = (_Float16)v;
            }
        }
    };

    v8f acc[2];
    acc[0] = (v8f){0.f,0.f,0.f,0.f,0.f,0.f,0.f,0.f};
    acc[1] = (v8f){0.f,0.f,0.f,0.f,0.f,0.f,0.f,0.f};

    const int m  = lane & 15;
    const int bc = wave * 16 + col;

    stage(0, 0);
    __syncthreads();

    int buf = 0;
    for (int kb = 0; kb < K; kb += TILE_K) {
        // prefetch the K+2 tile (global_prefetch_b8)
        if (kb + 2 * TILE_K < K)
            __builtin_prefetch(X + (size_t)(kb + 2 * TILE_K) * NP + p0 + (tid & 127), 0, 1);
        // stage next tile into the other buffer while computing this one
        if (kb + TILE_K < K) stage(buf ^ 1, kb + TILE_K);

        // fragments: two contiguous 8-half runs per lane (ds_load_b128):
        // lanes 0-15 hold K = 0..7 & 16..23, lanes 16-31 hold K = 8..15 & 24..31.
        v8h blo = *(const v8h*)&sxT[buf][bc][khalf];
        v8h bhi = *(const v8h*)&sxT[buf][bc][16 + khalf];
        v16h b  = __builtin_shufflevector(blo, bhi, 0,1,2,3,4,5,6,7,8,9,10,11,12,13,14,15);

        v8h a0l = *(const v8h*)&sa[buf][m][khalf];
        v8h a0h = *(const v8h*)&sa[buf][m][16 + khalf];
        v16h a0 = __builtin_shufflevector(a0l, a0h, 0,1,2,3,4,5,6,7,8,9,10,11,12,13,14,15);
        v8h a1l = *(const v8h*)&sa[buf][16 + m][khalf];
        v8h a1h = *(const v8h*)&sa[buf][16 + m][16 + khalf];
        v16h a1 = __builtin_shufflevector(a1l, a1h, 0,1,2,3,4,5,6,7,8,9,10,11,12,13,14,15);

        acc[0] = __builtin_amdgcn_wmma_f32_16x16x32_f16(false, a0, false, b,
                                                        (short)0, acc[0], false, false);
        acc[1] = __builtin_amdgcn_wmma_f32_16x16x32_f16(false, a1, false, b,
                                                        (short)0, acc[1], false, false);
        __syncthreads();
        buf ^= 1;
    }

    // epilogue; C/D layout: lane 0-15 -> col=lane, row=r; lane 16-31 -> row=r+8
    const int n = p0 + wave * 16 + col;
#pragma unroll
    for (int t = 0; t < 2; ++t) {
#pragma unroll
        for (int r = 0; r < 8; ++r) {
            int row = m0 + 16 * t + r + ((lane >> 4) << 3);
            if (row >= Cout) continue;
            float v = acc[t][r];
            if (bias) v += bias[row];
            if (wp)
                v += wp[row * 3 + 0] * xyzp[n] +
                     wp[row * 3 + 1] * xyzp[(size_t)NP + n] +
                     wp[row * 3 + 2] * xyzp[2 * (size_t)NP + n];
            if (bng) {
                float s = bng[row] * rsqrtf(bnv[row] + 1e-5f);
                v = (v - bnm[row]) * s + bnb[row];
            }
            if (act == 1)      v = fmaxf(v, 0.f);
            else if (act == 2) v = tanhf(v);
            if (resid) v += resid[(size_t)row * NP + n];
            if (out_mode == 0) {
                Y[(size_t)row * NP + n] = v;
            } else {  // (B, Cout, Nper) layout for final classifier output
                int bb = n / Nper, nn = n % Nper;
                Y[((size_t)bb * Cout + row) * Nper + nn] = v;
            }
        }
    }
}

// --------------------------- helper kernels --------------------------------
__global__ void pack_input(const float* __restrict__ in, float* __restrict__ x0,
                           float* __restrict__ xyz, float* __restrict__ dens,
                           int C, int Nper, int NP)
{
    int t = blockIdx.x * blockDim.x + threadIdx.x;
    if (t < 24) dens[t] = 0.f;                 // zero the density outputs
    int total = C * NP;
    if (t >= total) return;
    int ch = t / NP, pt = t % NP;
    int b = pt / Nper, n = pt % Nper;
    float v = in[((size_t)b * C + ch) * Nper + n];
    x0[(size_t)ch * NP + pt] = v;
    if (ch < 3) xyz[(size_t)ch * NP + pt] = v;
}

__global__ void zero_kernel(float* __restrict__ p, long n)
{
    long t = (long)blockIdx.x * blockDim.x + threadIdx.x;
    if (t < n) p[t] = 0.f;
}

__device__ __forceinline__
int make_corners(const float* __restrict__ keys, int h, int d, int S,
                 int NP, int pt, int* idx, float* w)
{
    int f0[3]; float tt[3];
#pragma unroll 3
    for (int ax = 0; ax < d; ++ax) {
        float k = keys[(size_t)(h * d + ax) * NP + pt];
        float g = (k + 1.f) * 0.5f * (float)(S - 1);
        g = fminf(fmaxf(g, 0.f), (float)(S - 1));
        int fi = (int)floorf(g);
        fi = min(max(fi, 0), S - 2);
        f0[ax] = fi; tt[ax] = g - (float)fi;
    }
    int nc = 1 << d;
    for (int c = 0; c < nc; ++c) {
        int id = 0; float ww = 1.f;
        for (int ax = 0; ax < d; ++ax) {
            int bit = (c >> (d - 1 - ax)) & 1;   // axis0-major flat index
            id = id * S + f0[ax] + bit;
            ww *= bit ? tt[ax] : (1.f - tt[ax]);
        }
        idx[c] = id; w[c] = ww;
    }
    return nc;
}

__global__ void splat_kernel(const float* __restrict__ keys,
                             const float* __restrict__ vals,
                             float* __restrict__ lat, float* __restrict__ wlat,
                             int H, int F, int d, int S, int size,
                             int NP, int Nper)
{
    int t = blockIdx.x * blockDim.x + threadIdx.x;
    int total = H * NP;
    if (t >= total) return;
    int pt = t % NP, h = t / NP;
    int b = pt / Nper, bh = b * H + h;
    int idx[8]; float w[8];
    int nc = make_corners(keys, h, d, S, NP, pt, idx, w);
    for (int c = 0; c < nc; ++c)
        atomicAdd(&wlat[(size_t)bh * size + idx[c]], w[c]);
    for (int f = 0; f < F; ++f) {
        float v = vals[(size_t)(h * F + f) * NP + pt];
        for (int c = 0; c < nc; ++c)
            atomicAdd(&lat[((size_t)bh * F + f) * size + idx[c]], v * w[c]);
    }
}

__global__ void norm_kernel(float* __restrict__ lat, const float* __restrict__ wlat,
                            int F, int size, long total)
{
    long t = (long)blockIdx.x * blockDim.x + threadIdx.x;
    if (t >= total) return;
    long cell = t % size;
    long bh   = (t / size) / F;
    lat[t] /= (wlat[bh * size + cell] + 0.001f);
}

__global__ void dens_kernel(const float* __restrict__ wlat, float* __restrict__ out,
                            long n, float inv)
{
    __shared__ float s[256];
    float v = 0.f;
    for (long i = (long)blockIdx.x * 256 + threadIdx.x; i < n;
         i += (long)gridDim.x * 256)
        v += wlat[i];
    s[threadIdx.x] = v;
    __syncthreads();
    for (int o = 128; o > 0; o >>= 1) {
        if (threadIdx.x < o) s[threadIdx.x] += s[threadIdx.x + o];
        __syncthreads();
    }
    if (threadIdx.x == 0) atomicAdd(out, s[0] * inv);
}

// grouped 3^d "SAME" conv (cross-correlation) + bias + relu on the lattice
__global__ void gconv_kernel(const float* __restrict__ lat,
                             const float* __restrict__ cw,
                             const float* __restrict__ cb,
                             float* __restrict__ act_out,
                             int B, int H, int F, int d, int S, int size)
{
    int t = blockIdx.x * blockDim.x + threadIdx.x;
    int total = B * H * F * size;
    if (t >= total) return;
    int cell = t % size, r = t / size;
    int fo = r % F; r /= F;
    int h = r % H, b = r / H;
    int a0, a1, a2 = 0;
    if (d == 2) { a0 = cell / S; a1 = cell % S; }
    else        { a0 = cell / (S * S); a1 = (cell / S) % S; a2 = cell % S; }
    int taps = (d == 2) ? 9 : 27;
    const float* wb = cw + (size_t)(h * F + fo) * F * taps;
    float acc = cb[h * F + fo];
    for (int fi = 0; fi < F; ++fi) {
        const float* wr = wb + (size_t)fi * taps;
        const float* lr = lat + ((size_t)(b * H + h) * F + fi) * size;
        if (d == 2) {
            int tap = 0;
            for (int o0 = -1; o0 <= 1; ++o0)
                for (int o1 = -1; o1 <= 1; ++o1, ++tap) {
                    int p0 = a0 + o0, p1 = a1 + o1;
                    if ((unsigned)p0 < (unsigned)S && (unsigned)p1 < (unsigned)S)
                        acc += wr[tap] * lr[p0 * S + p1];
                }
        } else {
            int tap = 0;
            for (int o0 = -1; o0 <= 1; ++o0)
                for (int o1 = -1; o1 <= 1; ++o1)
                    for (int o2 = -1; o2 <= 1; ++o2, ++tap) {
                        int p0 = a0 + o0, p1 = a1 + o1, p2 = a2 + o2;
                        if ((unsigned)p0 < (unsigned)S && (unsigned)p1 < (unsigned)S &&
                            (unsigned)p2 < (unsigned)S)
                            acc += wr[tap] * lr[(p0 * S + p1) * S + p2];
                    }
        }
    }
    act_out[t] = fmaxf(acc, 0.f);
}

__global__ void gather_kernel(const float* __restrict__ keys,
                              const float* __restrict__ act,
                              float* __restrict__ Y,
                              int H, int F, int d, int S, int size,
                              int NP, int Nper)
{
    int t = blockIdx.x * blockDim.x + threadIdx.x;
    int total = H * F * NP;
    if (t >= total) return;
    int pt = t % NP, ch = t / NP;
    int h = ch / F, f = ch % F;
    int b = pt / Nper, bh = b * H + h;
    int idx[8]; float w[8];
    int nc = make_corners(keys, h, d, S, NP, pt, idx, w);
    const float* ar = act + ((size_t)bh * F + f) * size;
    float s = 0.f;
    for (int c = 0; c < nc; ++c) s += ar[idx[c]] * w[c];
    Y[(size_t)ch * NP + pt] = s;
}

// ------------------------------ host side ----------------------------------
namespace {
struct Cfg { int H, F, S, d; };
static const Cfg BRCFG[3][2] = {
    { {16, 4, 128, 2}, {16, 4, 32, 3} },
    { {16,16,  64, 2}, {16,16, 16, 3} },
    { {16,16,  16, 2}, {16,32,  8, 3} },
};

inline void gemm(hipStream_t st, const float* W, const float* X, const float* bias,
                 const float* bng, const float* bnb, const float* bnm, const float* bnv,
                 const float* wp, const float* xyzp, const float* resid,
                 float* Y, int Cout, int K, int Kw, int NP, int act, int out_mode,
                 int Nper)
{
    dim3 g(NP / TILE_N, (Cout + TILE_M - 1) / TILE_M);
    wmma_gemm<<<g, 256, 0, st>>>(W, X, bias, bng, bnb, bnm, bnv,
                                 wp, xyzp, resid, Y, Cout, K, Kw, NP, act,
                                 out_mode, Nper);
}
inline void zero(hipStream_t st, float* p, long n)
{
    zero_kernel<<<(int)((n + 255) / 256), 256, 0, st>>>(p, n);
}
} // namespace

extern "C" void kernel_launch(void* const* d_in, const int* in_sizes, int n_in,
                              void* d_out, int out_size, void* d_ws, size_t ws_size,
                              hipStream_t stream)
{
    (void)in_sizes; (void)n_in; (void)out_size; (void)ws_size;
    const int B = 2, Nper = 8192, NP = B * Nper, M = 512, NCLS = 13;

    // ---- unpack pytree leaves (jax tree order: dict keys sorted) ----
    int li = 0;
    const float* input = (const float*)d_in[li++];

    struct BrP { const float *bk, *bv, *cb, *cw, *wk, *wp, *wv; };
    struct BlkP { const float *bnb, *bng, *bnm, *bnv, *bo; BrP br[2]; const float* wo; };
    BlkP blk[12];
    for (int i = 0; i < 12; ++i) {
        blk[i].bnb = (const float*)d_in[li++];   // bn.b
        blk[i].bng = (const float*)d_in[li++];   // bn.g
        blk[i].bnm = (const float*)d_in[li++];   // bn.m
        blk[i].bnv = (const float*)d_in[li++];   // bn.v
        blk[i].bo  = (const float*)d_in[li++];
        for (int j = 0; j < 2; ++j) {
            blk[i].br[j].bk = (const float*)d_in[li++];
            blk[i].br[j].bv = (const float*)d_in[li++];
            blk[i].br[j].cb = (const float*)d_in[li++];
            blk[i].br[j].cw = (const float*)d_in[li++];
            blk[i].br[j].wk = (const float*)d_in[li++];
            blk[i].br[j].wp = (const float*)d_in[li++];
            blk[i].br[j].wv = (const float*)d_in[li++];
        }
        blk[i].wo = (const float*)d_in[li++];
    }
    const float* f_b2  = (const float*)d_in[li++];
    const float* f_bnb = (const float*)d_in[li++];
    const float* f_bng = (const float*)d_in[li++];
    const float* f_bnm = (const float*)d_in[li++];
    const float* f_bnv = (const float*)d_in[li++];
    const float* f_w1  = (const float*)d_in[li++];
    const float* f_w2  = (const float*)d_in[li++];
    const float* s_b   = (const float*)d_in[li++];
    const float* s_bnb = (const float*)d_in[li++];
    const float* s_bng = (const float*)d_in[li++];
    const float* s_bnm = (const float*)d_in[li++];
    const float* s_bnv = (const float*)d_in[li++];
    const float* s_w   = (const float*)d_in[li++];

    // ---- workspace layout (floats) ----
    float* ws   = (float*)d_ws;
    float* x    = ws;                               // 512*NP
    float* xyz  = x    + (size_t)512 * NP;          // 3*NP
    float* x0   = xyz  + (size_t)3   * NP;          // 32*NP (K padded to 32)
    float* keys = x0   + (size_t)32  * NP;          // 48*NP
    float* vals = keys + (size_t)48  * NP;          // 512*NP
    float* cat  = vals + (size_t)512 * NP;          // 768*NP (reused as h)
    float* lat  = cat  + (size_t)768 * NP;          // 4,194,304
    float* actb = lat  + (size_t)4194304;           // 4,194,304
    float* wlat = actb + (size_t)4194304;           // 1,048,576

    float* res_out  = (float*)d_out;                          // (B,13,1,N)
    float* dens_out = (float*)d_out + (size_t)B * NCLS * Nper;// 24 scalars

    // ---- pack input (6 used rows), zero the 26 pad rows + dens slots ----
    {
        int tot = 6 * NP;
        pack_input<<<(tot + 255) / 256, 256, 0, stream>>>(input, x0, xyz, dens_out,
                                                          6, Nper, NP);
        zero(stream, x0 + (size_t)6 * NP, (long)26 * NP);
    }
    // ---- stem: x = relu(bn(W x0 + b)), K padded to 32, Kw = 6 ----
    gemm(stream, s_w, x0, s_b, s_bng, s_bnb, s_bnm, s_bnv,
         nullptr, nullptr, nullptr, x, M, TILE_K, 6, NP, /*act=*/1, 0, Nper);

    int dens_i = 0;
    for (int bi = 0; bi < 12; ++bi) {
        const Cfg* cf = BRCFG[bi % 3];
        int cbase = 0;
        for (int j = 0; j < 2; ++j) {
            const Cfg c = cf[j];
            const BrP& p = blk[bi].br[j];
            int size = 1; for (int q = 0; q < c.d; ++q) size *= c.S;
            const int HD = c.H * c.d, HF = c.H * c.F;

            // keys = tanh(Wk x + bk + Wp xyz)
            gemm(stream, p.wk, x, p.bk, nullptr, nullptr, nullptr, nullptr,
                 p.wp, xyz, nullptr, keys, HD, M, M, NP, /*act=*/2, 0, Nper);
            // vals = Wv x + bv
            gemm(stream, p.wv, x, p.bv, nullptr, nullptr, nullptr, nullptr,
                 nullptr, nullptr, nullptr, vals, HF, M, M, NP, /*act=*/0, 0, Nper);

            long latN  = (long)B * c.H * c.F * size;
            long wlatN = (long)B * c.H * size;
            zero(stream, lat,  latN);
            zero(stream, wlat, wlatN);

            int tot = c.H * NP;
            splat_kernel<<<(tot + 255) / 256, 256, 0, stream>>>(
                keys, vals, lat, wlat, c.H, c.F, c.d, c.S, size, NP, Nper);
            norm_kernel<<<(int)((latN + 255) / 256), 256, 0, stream>>>(
                lat, wlat, c.F, size, latN);
            dens_kernel<<<256, 256, 0, stream>>>(wlat, dens_out + dens_i,
                                                 wlatN, 1.f / (float)wlatN);
            ++dens_i;

            int ctot = B * c.H * c.F * size;
            gconv_kernel<<<(ctot + 255) / 256, 256, 0, stream>>>(
                lat, p.cw, p.cb, actb, B, c.H, c.F, c.d, c.S, size);

            int gtot = HF * NP;
            gather_kernel<<<(gtot + 255) / 256, 256, 0, stream>>>(
                keys, actb, cat + (size_t)cbase * NP,
                c.H, c.F, c.d, c.S, size, NP, Nper);
            cbase += HF;
        }
        // x = x + relu(bn(Wo cat + bo))   (in place: residual read == write site)
        gemm(stream, blk[bi].wo, cat, blk[bi].bo,
             blk[bi].bng, blk[bi].bnb, blk[bi].bnm, blk[bi].bnv,
             nullptr, nullptr, /*resid=*/x, x, M, cbase, cbase, NP, /*act=*/1, 0, Nper);
    }

    // ---- final head ----
    float* hbuf = cat;  // cat is free now
    gemm(stream, f_w1, x, nullptr, f_bng, f_bnb, f_bnm, f_bnv,
         nullptr, nullptr, nullptr, hbuf, M, M, M, NP, /*act=*/1, 0, Nper);
    gemm(stream, f_w2, hbuf, f_b2, nullptr, nullptr, nullptr, nullptr,
         nullptr, nullptr, nullptr, res_out, NCLS, M, M, NP, /*act=*/0,
         /*out_mode=*/1, Nper);
}
